// VQGAN_70360154243638
// MI455X (gfx1250) — compile-verified, matrix operand order checked by reading, not run on hardware
//
#include <hip/hip_runtime.h>
#include <hip/hip_bf16.h>
#include <math.h>

// ---------------------------------------------------------------------------
// CDNA5 (gfx1250) VQGAN forward. All contractions (3x3/1x1 convs as implicit
// GEMM, attention QK^T and V*W) run on v_wmma_f32_16x16x32_bf16, wave32.
// Conv: one wave per 16(M=pixels) x 64(N=cout) tile, 4 accumulators sharing
// one A fragment. Weights pre-transposed+converted to bf16 rows (zero-padded
// in both K and Cout) so B fragments are aligned uint4 loads with no guards.
// The 3-channel stem input is zero-padded to 32 channels so every conv takes
// the division-free shift/mask fast path.
// ---------------------------------------------------------------------------

typedef __attribute__((ext_vector_type(16))) __bf16 v16bf;
typedef __attribute__((ext_vector_type(8)))  float  v8f;

union V16U {
    v16bf v;
    __bf16 e[16];
    unsigned short u[16];
    uint4 q[2];
};

__device__ __forceinline__ unsigned short bfbits(float f) {
    union { __bf16 b; unsigned short u; } c;
    c.b = (__bf16)f;
    return c.u;
}

// K-index mapping for 16-bit A/B operands (ISA 7.12.2):
// lanes 0-15: e<8 -> K=e,    e>=8 -> K=e+8   (K 0..7, 16..23)
// lanes16-31: e<8 -> K=e+8,  e>=8 -> K=e+16  (K 8..15, 24..31)
__device__ __forceinline__ int kmap(int e, int half) {
    return (e < 8) ? (e + 8 * half) : (e + 8 + 8 * half);
}

__device__ __forceinline__ void cvt16(V16U& d, float4 a0, float4 a1,
                                      float4 a2, float4 a3) {
    d.e[0]  = (__bf16)a0.x; d.e[1]  = (__bf16)a0.y;
    d.e[2]  = (__bf16)a0.z; d.e[3]  = (__bf16)a0.w;
    d.e[4]  = (__bf16)a1.x; d.e[5]  = (__bf16)a1.y;
    d.e[6]  = (__bf16)a1.z; d.e[7]  = (__bf16)a1.w;
    d.e[8]  = (__bf16)a2.x; d.e[9]  = (__bf16)a2.y;
    d.e[10] = (__bf16)a2.z; d.e[11] = (__bf16)a2.w;
    d.e[12] = (__bf16)a3.x; d.e[13] = (__bf16)a3.y;
    d.e[14] = (__bf16)a3.z; d.e[15] = (__bf16)a3.w;
}

// ---------------------------------------------------------------------------
// Stem input channel pad: [pix,3] -> [pix,32] (zero fill).
// ---------------------------------------------------------------------------
__global__ void k_padchan(const float* __restrict__ x, float* __restrict__ y,
                          long n)
{
    long i = (long)blockIdx.x * blockDim.x + threadIdx.x;
    if (i >= n) return;
    const int c = (int)(i & 31);
    const long pix = i >> 5;
    y[i] = (c < 3) ? x[pix * 3 + c] : 0.f;
}

// ---------------------------------------------------------------------------
// Weight transpose + bf16 convert + pad:
// wt[co][kk*CinP + ci] = bf16(w[(kk*Cin+ci)*Cout + co]), zero for ci>=Cin or
// co>=Cout. KtotP = KH*KW*CinP (CinP multiple of 32).
// ---------------------------------------------------------------------------
__global__ __launch_bounds__(256) void k_wtrans(
    const float* __restrict__ w, unsigned short* __restrict__ wt,
    int Cin, int cinPShift, int Cout, int KtotP)
{
    const int co = blockIdx.y;
    const int k  = blockIdx.x * blockDim.x + threadIdx.x;
    if (k >= KtotP) return;
    const int kk = k >> cinPShift;
    const int ci = k & ((1 << cinPShift) - 1);
    const float v = (ci < Cin && co < Cout)
                  ? w[(size_t)(kk * Cin + ci) * Cout + co] : 0.f;
    wt[(size_t)co * KtotP + k] = bfbits(v);
}

// ---------------------------------------------------------------------------
// Implicit-GEMM conv2d (NHWC in with CinP channels, bf16-transposed weights).
// One wave computes a 16(M) x 64(N) tile: 4 accumulators share one A
// fragment. Square spatial, Wo = Ho = power of two. CinP = power of two.
// KW is a template constant (1 or 3) so all index math is shifts/magic-mults.
// ---------------------------------------------------------------------------
template <int KWc>
__global__ __launch_bounds__(32) void k_conv2d_wmma(
    const float* __restrict__ x, const unsigned short* __restrict__ wt,
    const float* __restrict__ bias, float* __restrict__ y,
    int H, int W, int CinP, int Cout,
    int stride, int pad, int Wo, int hwShift, int KtotP, int cinShift)
{
    const int lane = threadIdx.x;
    const int half = lane >> 4;
    const int lm   = lane & 15;
    const int m0   = blockIdx.x * 16;
    const int n0   = blockIdx.y * 64;

    // A-operand row (output pixel): power-of-two decompose
    const int mA  = m0 + lm;
    const int wo  = mA & (Wo - 1);
    const int t1  = mA >> hwShift;
    const int ho  = t1 & (Wo - 1);           // Ho == Wo (square)
    const int bat = t1 >> hwShift;

    // B rows (transposed bf16 weights, contiguous in k; padded -> no guards)
    const unsigned short* w0 = wt + (size_t)(n0 + lm) * KtotP + 8 * half;
    const size_t wstep = (size_t)16 * KtotP;

    const int cmask = CinP - 1;

    v8f acc0 = {}, acc1 = {}, acc2 = {}, acc3 = {};

#pragma unroll 2
    for (int k0 = 0; k0 < KtotP; k0 += 32) {
        const int kk  = k0 >> cinShift;       // ky*KW + kx (uniform scalar)
        const int ci0 = k0 & cmask;
        const int ky  = kk / KWc;             // compile-time divisor
        const int kx  = kk - ky * KWc;
        const int hi  = ho * stride + ky - pad;
        const int wi  = wo * stride + kx - pad;
        const bool inb = ((unsigned)hi < (unsigned)H) &&
                         ((unsigned)wi < (unsigned)W);
        const int hic = inb ? hi : 0;
        const int wic = inb ? wi : 0;
        const float* ax = x + ((size_t)((bat * H + hic) * W + wic)) * CinP
                            + ci0 + 8 * half;
        const float4 z4 = make_float4(0.f, 0.f, 0.f, 0.f);
        const float4 a0 = inb ? *(const float4*)(ax)      : z4;
        const float4 a1 = inb ? *(const float4*)(ax + 4)  : z4;
        const float4 a2 = inb ? *(const float4*)(ax + 16) : z4;
        const float4 a3 = inb ? *(const float4*)(ax + 20) : z4;
        V16U Am;
        cvt16(Am, a0, a1, a2, a3);

        V16U B0, B1, B2, B3;
        const unsigned short* wp = w0 + k0;
        B0.q[0] = *(const uint4*)(wp);
        B0.q[1] = *(const uint4*)(wp + 16);
        B1.q[0] = *(const uint4*)(wp + wstep);
        B1.q[1] = *(const uint4*)(wp + wstep + 16);
        B2.q[0] = *(const uint4*)(wp + 2 * wstep);
        B2.q[1] = *(const uint4*)(wp + 2 * wstep + 16);
        B3.q[0] = *(const uint4*)(wp + 3 * wstep);
        B3.q[1] = *(const uint4*)(wp + 3 * wstep + 16);

        acc0 = __builtin_amdgcn_wmma_f32_16x16x32_bf16(
            false, Am.v, false, B0.v, (short)0, acc0, false, false);
        acc1 = __builtin_amdgcn_wmma_f32_16x16x32_bf16(
            false, Am.v, false, B1.v, (short)0, acc1, false, false);
        acc2 = __builtin_amdgcn_wmma_f32_16x16x32_bf16(
            false, Am.v, false, B2.v, (short)0, acc2, false, false);
        acc3 = __builtin_amdgcn_wmma_f32_16x16x32_bf16(
            false, Am.v, false, B3.v, (short)0, acc3, false, false);
    }

    // C/D layout: lane holds column N = lane%16, rows M = r + 8*(lane/16)
#pragma unroll
    for (int tix = 0; tix < 4; tix++) {
        const int n = n0 + 16 * tix + lm;
        if (n < Cout) {
            const v8f& a = (tix == 0) ? acc0 : (tix == 1) ? acc1
                         : (tix == 2) ? acc2 : acc3;
            const float bb = bias[n];
#pragma unroll
            for (int r = 0; r < 8; r++)
                y[(size_t)(m0 + r + 8 * half) * Cout + n] = a[r] + bb;
        }
    }
}

// ---------------------------------------------------------------------------
// Batched NT GEMM: C[b,m,n] = scale * sum_k A[b,m,k] * B[b,n,k]
// (attention scores: A=q, B=k, M=N=P, K=C). 16x32 tile, 2 accumulators.
// ---------------------------------------------------------------------------
__global__ __launch_bounds__(32) void k_gemm_nt_wmma(
    const float* __restrict__ A, const float* __restrict__ B,
    float* __restrict__ C, int M, int N, int K, float scale)
{
    const int lane = threadIdx.x, half = lane >> 4, lm = lane & 15;
    const size_t bat = blockIdx.z;
    const float* Ab = A + bat * (size_t)M * K;
    const float* Bb = B + bat * (size_t)N * K;
    float* Cb = C + bat * (size_t)M * N;
    const int m0 = blockIdx.x * 16, n0 = blockIdx.y * 32;
    const float* ar  = Ab + (size_t)(m0 + lm) * K + 8 * half;
    const float* br0 = Bb + (size_t)(n0 + lm) * K + 8 * half;
    const float* br1 = br0 + (size_t)16 * K;

    v8f acc0 = {}, acc1 = {};
#pragma unroll 2
    for (int k0 = 0; k0 < K; k0 += 32) {
        V16U Am, B0, B1;
        cvt16(Am, *(const float4*)(ar + k0),      *(const float4*)(ar + k0 + 4),
                  *(const float4*)(ar + k0 + 16), *(const float4*)(ar + k0 + 20));
        cvt16(B0, *(const float4*)(br0 + k0),      *(const float4*)(br0 + k0 + 4),
                  *(const float4*)(br0 + k0 + 16), *(const float4*)(br0 + k0 + 20));
        cvt16(B1, *(const float4*)(br1 + k0),      *(const float4*)(br1 + k0 + 4),
                  *(const float4*)(br1 + k0 + 16), *(const float4*)(br1 + k0 + 20));
        acc0 = __builtin_amdgcn_wmma_f32_16x16x32_bf16(
            false, Am.v, false, B0.v, (short)0, acc0, false, false);
        acc1 = __builtin_amdgcn_wmma_f32_16x16x32_bf16(
            false, Am.v, false, B1.v, (short)0, acc1, false, false);
    }
#pragma unroll
    for (int r = 0; r < 8; r++) {
        Cb[(size_t)(m0 + r + 8 * half) * N + n0 + lm]      = acc0[r] * scale;
        Cb[(size_t)(m0 + r + 8 * half) * N + n0 + 16 + lm] = acc1[r] * scale;
    }
}

// ---------------------------------------------------------------------------
// Batched TN GEMM: C[b,m,n] = sum_k A[b,k,m] * B[b,k,n]
// (attention apply: A=v [P,C] -> m over C, B=w [P,P] -> n over P, k over P).
// Output is [B, C, P] contiguous (then raw-reshaped to NHWC, faithful quirk).
// ---------------------------------------------------------------------------
__global__ __launch_bounds__(32) void k_gemm_tn_wmma(
    const float* __restrict__ A, const float* __restrict__ B,
    float* __restrict__ C, int M, int N, int K)
{
    const int lane = threadIdx.x, half = lane >> 4, lm = lane & 15;
    const size_t bat = blockIdx.z;
    const float* Ab = A + bat * (size_t)K * M;
    const float* Bb = B + bat * (size_t)K * N;
    float* Cb = C + bat * (size_t)M * N;
    const int m0 = blockIdx.x * 16, n0 = blockIdx.y * 16;
    const int m = m0 + lm, n = n0 + lm;

    v8f acc = {};
    for (int k0 = 0; k0 < K; k0 += 32) {
        V16U Am, Bm;
#pragma unroll
        for (int e = 0; e < 16; e++) {
            const int k = k0 + kmap(e, half);
            Am.e[e] = (__bf16)Ab[(size_t)k * M + m];
            Bm.e[e] = (__bf16)Bb[(size_t)k * N + n];
        }
        acc = __builtin_amdgcn_wmma_f32_16x16x32_bf16(
            false, Am.v, false, Bm.v, (short)0, acc, false, false);
    }
#pragma unroll
    for (int r = 0; r < 8; r++)
        Cb[(size_t)(m0 + r + 8 * half) * N + n0 + lm] = acc[r];
}

// ---------------------------------------------------------------------------
// GroupNorm (32 groups), optional fused swish. One block per (batch, group).
// cpg = C/32 is a power of two (4 or 8) -> shift/mask indexing.
// ---------------------------------------------------------------------------
__global__ __launch_bounds__(256) void k_groupnorm(
    const float* __restrict__ x, float* __restrict__ y,
    const float* __restrict__ gamma, const float* __restrict__ beta,
    int HW, int C, int cpgShift, int doSwish)
{
    __shared__ float s1[256], s2[256];
    const int tid = threadIdx.x;
    const int g = blockIdx.x & 31;
    const int b = blockIdx.x >> 5;
    const int cpg = 1 << cpgShift;
    const float* xb = x + (size_t)b * HW * C;
    float* yb = y + (size_t)b * HW * C;
    const int nred = HW << cpgShift;

    float sum = 0.f, ss = 0.f;
    for (int i = tid; i < nred; i += blockDim.x) {
        const int pix = i >> cpgShift;
        const int cc  = i & (cpg - 1);
        const float v = xb[(size_t)pix * C + (g << cpgShift) + cc];
        sum += v; ss += v * v;
    }
    s1[tid] = sum; s2[tid] = ss;
    __syncthreads();
    for (int s = 128; s > 0; s >>= 1) {
        if (tid < s) { s1[tid] += s1[tid + s]; s2[tid] += s2[tid + s]; }
        __syncthreads();
    }
    const float mean = s1[0] / (float)nred;
    const float var  = s2[0] / (float)nred - mean * mean;
    const float inv  = rsqrtf(var + 1e-6f);

    for (int i = tid; i < nred; i += blockDim.x) {
        const int pix = i >> cpgShift;
        const int cc  = i & (cpg - 1);
        const int c   = (g << cpgShift) + cc;
        float v = (xb[(size_t)pix * C + c] - mean) * inv;
        v = v * gamma[c] + beta[c];
        if (doSwish) v = v / (1.f + __expf(-v));
        yb[(size_t)pix * C + c] = v;
    }
}

// ---------------------------------------------------------------------------
// Row softmax, in place. One block per row, P <= 256.
// ---------------------------------------------------------------------------
__global__ __launch_bounds__(256) void k_softmax_rows(float* __restrict__ s, int P)
{
    __shared__ float red[256];
    float* r = s + (size_t)blockIdx.x * P;
    const int tid = threadIdx.x;

    float mx = -3.0e38f;
    for (int i = tid; i < P; i += blockDim.x) mx = fmaxf(mx, r[i]);
    red[tid] = mx;
    __syncthreads();
    for (int st = 128; st > 0; st >>= 1) {
        if (tid < st) red[tid] = fmaxf(red[tid], red[tid + st]);
        __syncthreads();
    }
    mx = red[0];
    __syncthreads();

    float sum = 0.f;
    for (int i = tid; i < P; i += blockDim.x) {
        const float e = __expf(r[i] - mx);
        r[i] = e; sum += e;
    }
    red[tid] = sum;
    __syncthreads();
    for (int st = 128; st > 0; st >>= 1) {
        if (tid < st) red[tid] += red[tid + st];
        __syncthreads();
    }
    const float is = 1.f / red[0];
    for (int i = tid; i < P; i += blockDim.x) r[i] *= is;
}

// ---------------------------------------------------------------------------
// Elementwise helpers
// ---------------------------------------------------------------------------
__global__ void k_add(const float* __restrict__ a, const float* __restrict__ b,
                      float* __restrict__ o, long n)
{
    long i = (long)blockIdx.x * blockDim.x + threadIdx.x;
    if (i < n) o[i] = a[i] + b[i];
}

__global__ void k_sigmoid(const float* __restrict__ a, float* __restrict__ o, long n)
{
    long i = (long)blockIdx.x * blockDim.x + threadIdx.x;
    if (i < n) o[i] = 1.f / (1.f + __expf(-a[i]));
}

// nearest 2x upsample; C, 2W, 2H all powers of two -> shift/mask
__global__ void k_upsample2x(const float* __restrict__ x, float* __restrict__ y,
                             int H, int W, int cShift, long n)
{
    long i = (long)blockIdx.x * blockDim.x + threadIdx.x;
    if (i >= n) return;
    const int C = 1 << cShift;
    const int c = (int)(i & (C - 1));
    long t = i >> cShift;
    const int w2 = (int)(t & (2 * W - 1)); t >>= (31 - __clz(2 * W));
    const int h2 = (int)(t & (2 * H - 1)); t >>= (31 - __clz(2 * H));
    const int b  = (int)t;
    y[i] = x[(size_t)(((b * H + (h2 >> 1)) * W + (w2 >> 1))) << cShift | c];
}

// ---------------------------------------------------------------------------
// Nearest-codebook quantize: zq[v] = codebook[argmin_j ||z[v]-c[j]||^2]
// ---------------------------------------------------------------------------
__global__ __launch_bounds__(256) void k_quantize(
    const float* __restrict__ z, const float* __restrict__ cb,
    float* __restrict__ zq, int NV, int Cd, int NC)
{
    const int v = blockIdx.x * blockDim.x + threadIdx.x;
    if (v >= NV) return;
    const float* zz = z + (size_t)v * Cd;
    int best = 0; float bd = 3.0e38f;
    for (int j = 0; j < NC; j++) {
        const float* cj = cb + (size_t)j * Cd;
        float d = 0.f;
        for (int c = 0; c < Cd; c++) { const float t = zz[c] - cj[c]; d += t * t; }
        if (d < bd) { bd = d; best = j; }
    }
    const float* cj = cb + (size_t)best * Cd;
    float* o = zq + (size_t)v * Cd;
    for (int c = 0; c < Cd; c++) o[c] = cj[c];
}

// ---------------------------------------------------------------------------
// Host orchestration
// ---------------------------------------------------------------------------
struct ConvP { const float* w; const float* b; int kh, kw, cin, cout; };
struct GNP   { const float* g; const float* b; };
struct ResP  { GNP n1; ConvP c1; GNP n2; ConvP c2; ConvP nin; bool has_nin; };
struct AttnP { GNP n; ConvP q, k, v, o; };

extern "C" void kernel_launch(void* const* d_in, const int* in_sizes, int n_in,
                              void* d_out, int out_size, void* d_ws, size_t ws_size,
                              hipStream_t stream)
{
    (void)in_sizes; (void)n_in; (void)out_size;
    const int Bn = 32;

    // ---- unpack parameters (setup_inputs() dict insertion order) ----
    int idx = 0;
    auto nf = [&]() -> const float* { return (const float*)d_in[idx++]; };
    auto getConv = [&](int kh, int kw, int cin, int cout) {
        ConvP c; c.w = nf(); c.b = nf();
        c.kh = kh; c.kw = kw; c.cin = cin; c.cout = cout; return c;
    };
    auto getGN = [&]() { GNP g; g.g = nf(); g.b = nf(); return g; };
    auto getRes = [&](int cin, int cout) {
        ResP r;
        r.n1 = getGN(); r.c1 = getConv(3, 3, cin, cout);
        r.n2 = getGN(); r.c2 = getConv(3, 3, cout, cout);
        r.has_nin = (cin != cout);
        if (r.has_nin) r.nin = getConv(1, 1, cin, cout); else r.nin = r.c2;
        return r;
    };
    auto getAttn = [&](int c) {
        AttnP a; a.n = getGN();
        a.q = getConv(1, 1, c, c); a.k = getConv(1, 1, c, c);
        a.v = getConv(1, 1, c, c); a.o = getConv(1, 1, c, c);
        return a;
    };

    const float* x_in = nf();                               // [32,64,64,3]
    // encoder
    ConvP e_conv_in = getConv(3, 3, 3, 128);
    ResP  e_r0 = getRes(128, 128); ConvP e_d0 = getConv(3, 3, 128, 128);
    ResP  e_r1 = getRes(128, 128); ConvP e_d1 = getConv(3, 3, 128, 128);
    ResP  e_r2 = getRes(128, 256); ConvP e_d2 = getConv(3, 3, 256, 256);
    ResP  e_r3 = getRes(256, 256);
    ResP  e_mb1 = getRes(256, 256); AttnP e_attn = getAttn(256);
    ResP  e_mb2 = getRes(256, 256);
    GNP   e_nout = getGN(); ConvP e_cout = getConv(3, 3, 256, 128);
    // decoder
    ConvP d_conv_in = getConv(3, 3, 128, 256);
    ResP  d_mb1 = getRes(256, 256); AttnP d_attn = getAttn(256);
    ResP  d_mb2 = getRes(256, 256);
    ResP  d_u0 = getRes(256, 256); ResP d_u1 = getRes(256, 256);
    ConvP d_up0 = getConv(3, 3, 256, 256);
    ResP  d_u2 = getRes(256, 256); AttnP d_a2a = getAttn(256);
    ResP  d_u3 = getRes(256, 256); AttnP d_a2b = getAttn(256);
    ConvP d_up1 = getConv(3, 3, 256, 256);
    ResP  d_u4 = getRes(256, 128); ResP d_u5 = getRes(128, 128);
    ConvP d_up2 = getConv(3, 3, 128, 128);
    ResP  d_u6 = getRes(128, 128); ResP d_u7 = getRes(128, 128);
    GNP   d_nout = getGN(); ConvP d_cout = getConv(3, 3, 128, 3);
    const float* codebook = nf();                           // [1024,128]

    // ---- workspace carve ----
    char* wp = (char*)d_ws;
    auto carve = [&](size_t nbytes) -> char* {
        char* p = wp;
        wp += ((nbytes + 255) / 256) * 256;
        return p;
    };
    const size_t MAXE = (size_t)32 * 64 * 64 * 128;   // 16.7M floats / buffer
    const size_t PE   = (size_t)32 * 256 * 256;       // attention buffers
    const size_t WTE  = (size_t)256 * 2304;           // max transposed weight
    const size_t XPE  = (size_t)32 * 64 * 64 * 32;    // padded stem input
    const size_t need = 4 * ((MAXE * 4 + 255) / 256) * 256
                      + 5 * ((PE * 4 + 255) / 256) * 256
                      + ((WTE * 2 + 255) / 256) * 256
                      + ((XPE * 4 + 255) / 256) * 256;
    if (ws_size < need) return;
    float* bufA = (float*)carve(MAXE * 4);
    float* bufB = (float*)carve(MAXE * 4);
    float* bufC = (float*)carve(MAXE * 4);
    float* bufD = (float*)carve(MAXE * 4);
    float* qb = (float*)carve(PE * 4);
    float* kb = (float*)carve(PE * 4);
    float* vb = (float*)carve(PE * 4);
    float* sb = (float*)carve(PE * 4);
    float* hb = (float*)carve(PE * 4);
    unsigned short* wbuf = (unsigned short*)carve(WTE * 2);
    float* xpad = (float*)carve(XPE * 4);

    // ---- launch helpers ----
    // `in` has CinP channels (CinP = Cin except for the padded stem).
    auto convLP = [&](const float* in, float* out, const ConvP& c,
                      int H, int W, int stride, int pad, int CinP) {
        const int Ho = H / stride, Wo = W / stride;    // SAME, even dims
        const int KtotP = c.kh * c.kw * CinP;          // multiple of 32
        const int CoutP = (c.cout + 63) & ~63;
        const int cinPShift = __builtin_ctz(CinP);
        dim3 tg((KtotP + 255) / 256, CoutP);
        k_wtrans<<<tg, dim3(256), 0, stream>>>(c.w, wbuf, c.cin, cinPShift,
                                               c.cout, KtotP);
        const int hwShift = __builtin_ctz(Wo);
        dim3 grid(Bn * Ho * Wo / 16, CoutP / 64, 1);
        if (c.kw == 3)
            k_conv2d_wmma<3><<<grid, dim3(32), 0, stream>>>(
                in, wbuf, c.b, out, H, W, CinP, c.cout,
                stride, pad, Wo, hwShift, KtotP, cinPShift);
        else
            k_conv2d_wmma<1><<<grid, dim3(32), 0, stream>>>(
                in, wbuf, c.b, out, H, W, CinP, c.cout,
                stride, pad, Wo, hwShift, KtotP, cinPShift);
    };
    auto convL = [&](const float* in, float* out, const ConvP& c,
                     int H, int W, int stride, int pad) {
        convLP(in, out, c, H, W, stride, pad, c.cin);
    };
    auto gnL = [&](const float* in, float* out, const GNP& g,
                   int H, int W, int C, int sw) {
        const int cpgShift = __builtin_ctz(C / 32);
        k_groupnorm<<<dim3(Bn * 32), dim3(256), 0, stream>>>(
            in, out, g.g, g.b, H * W, C, cpgShift, sw);
    };
    auto addL = [&](const float* a, const float* b, float* o, long n) {
        k_add<<<dim3((unsigned)((n + 255) / 256)), dim3(256), 0, stream>>>(a, b, o, n);
    };

    float *cur, *f1, *f2, *f3;

    auto resL = [&](const ResP& r, int H, int W) {
        gnL(cur, f1, r.n1, H, W, r.c1.cin, 1);
        convL(f1, f2, r.c1, H, W, 1, 1);
        gnL(f2, f1, r.n2, H, W, r.c1.cout, 1);
        convL(f1, f3, r.c2, H, W, 1, 1);
        const float* sc = cur;
        if (r.has_nin) { convL(cur, f2, r.nin, H, W, 1, 0); sc = f2; }
        addL(sc, f3, f1, (long)Bn * H * W * r.c1.cout);
        float* t = cur; cur = f1; f1 = f2; f2 = f3; f3 = t;
    };
    auto attnL = [&](const AttnP& a, int H, int W) {
        const int C = 256, P = H * W;
        gnL(cur, f1, a.n, H, W, C, 0);
        convL(f1, qb, a.q, H, W, 1, 0);
        convL(f1, kb, a.k, H, W, 1, 0);
        convL(f1, vb, a.v, H, W, 1, 0);
        dim3 gs(P / 16, P / 32, Bn);
        k_gemm_nt_wmma<<<gs, dim3(32), 0, stream>>>(
            qb, kb, sb, P, P, C, 1.0f / sqrtf((float)C));
        k_softmax_rows<<<dim3(Bn * P), dim3(256), 0, stream>>>(sb, P);
        dim3 gt(C / 16, P / 16, Bn);
        k_gemm_tn_wmma<<<gt, dim3(32), 0, stream>>>(vb, sb, hb, C, P, P);
        convL(hb, f1, a.o, H, W, 1, 0);  // hb raw-reshaped [B,C,P]->NHWC (quirk)
        addL(cur, f1, f2, (long)Bn * P * C);
        float* t = cur; cur = f2; f2 = t;
    };
    auto upL = [&](const ConvP& c, int H, int W, int C) {
        long n = (long)Bn * 2 * H * 2 * W * C;
        k_upsample2x<<<dim3((unsigned)((n + 255) / 256)), dim3(256), 0, stream>>>(
            cur, f1, H, W, __builtin_ctz(C), n);
        convL(f1, f2, c, 2 * H, 2 * W, 1, 1);
        float* t = cur; cur = f2; f2 = t;
    };
    auto downL = [&](const ConvP& c, int H, int W) {
        convL(cur, f1, c, H, W, 2, 0);   // SAME stride-2: pad_lo = 0
        float* t = cur; cur = f1; f1 = t;
    };

    // ---- encoder ----
    cur = bufA; f1 = bufB; f2 = bufC; f3 = bufD;
    {   // stem: pad 3 -> 32 channels, then division-free fast conv
        long n = (long)XPE;
        k_padchan<<<dim3((unsigned)((n + 255) / 256)), dim3(256), 0, stream>>>(
            x_in, xpad, n);
        convLP(xpad, cur, e_conv_in, 64, 64, 1, 1, 32);
    }
    resL(e_r0, 64, 64); downL(e_d0, 64, 64);          // -> 32x32
    resL(e_r1, 32, 32); downL(e_d1, 32, 32);          // -> 16x16
    resL(e_r2, 16, 16); downL(e_d2, 16, 16);          // -> 8x8
    resL(e_r3, 8, 8);
    resL(e_mb1, 8, 8); attnL(e_attn, 8, 8); resL(e_mb2, 8, 8);
    gnL(cur, f1, e_nout, 8, 8, 256, 1);
    convL(f1, f2, e_cout, 8, 8, 1, 1);                // z: [32,8,8,128] in f2

    // ---- quantize (forward value of straight-through is zq) ----
    k_quantize<<<dim3((2048 + 255) / 256), dim3(256), 0, stream>>>(
        f2, codebook, f3, 2048, 128, 1024);           // zq in f3
    { float* t = cur; cur = f3; f3 = t; }

    // ---- decoder ----
    convL(cur, f1, d_conv_in, 8, 8, 1, 1);
    { float* t = cur; cur = f1; f1 = t; }
    resL(d_mb1, 8, 8); attnL(d_attn, 8, 8); resL(d_mb2, 8, 8);
    resL(d_u0, 8, 8); resL(d_u1, 8, 8);
    upL(d_up0, 8, 8, 256);                            // -> 16x16
    resL(d_u2, 16, 16); attnL(d_a2a, 16, 16);
    resL(d_u3, 16, 16); attnL(d_a2b, 16, 16);
    upL(d_up1, 16, 16, 256);                          // -> 32x32
    resL(d_u4, 32, 32); resL(d_u5, 32, 32);
    upL(d_up2, 32, 32, 128);                          // -> 64x64
    resL(d_u6, 64, 64); resL(d_u7, 64, 64);
    gnL(cur, f1, d_nout, 64, 64, 128, 1);
    convL(f1, f2, d_cout, 64, 64, 1, 1);              // [32,64,64,3]
    {
        long n = (long)Bn * 64 * 64 * 3;
        k_sigmoid<<<dim3((unsigned)((n + 255) / 256)), dim3(256), 0, stream>>>(
            f2, (float*)d_out, n);
    }
}
